// Attention_8340826489010
// MI455X (gfx1250) — matrix operand order repaired and checked by reading, not compiled, not run on hardware
//
#include <hip/hip_runtime.h>
#include <hip/hip_bf16.h>

typedef __attribute__((ext_vector_type(16))) __bf16 v16bf;
typedef __attribute__((ext_vector_type(8)))  __bf16 bf16x8;
typedef __attribute__((ext_vector_type(8)))  float  v8f;

union Frag16 { v16bf v; bf16x8 h[2]; };

#define WMMA_BF16(a, b, c) \
  __builtin_amdgcn_wmma_f32_16x16x32_bf16(false, (a), false, (b), (short)0, (c), false, false)

// Problem constants (match the reference)
constexpr int BATCH = 2;
constexpr int NSEQ  = 2048;
constexpr int DIM   = 1024;
constexpr int NH    = 16;
constexpr int HD    = 64;          // head dim
constexpr float QSCALE = 0.125f;   // HD^-0.5 = 1/8
constexpr int ROWS  = BATCH * NSEQ;        // 4096
constexpr int QKVC  = 3 * DIM;             // 3072

// ---------------------------------------------------------------------------
// fp32 -> bf16 elementwise convert (grid-stride)
// ---------------------------------------------------------------------------
__global__ void attn_f2bf_kernel(const float* __restrict__ in,
                                 __bf16* __restrict__ out, int n) {
  int i = blockIdx.x * blockDim.x + threadIdx.x;
  int stride = gridDim.x * blockDim.x;
  for (; i < n; i += stride) out[i] = (__bf16)in[i];
}

// ---------------------------------------------------------------------------
// Fused fp32->bf16 transpose: W[K][N] (fp32) -> WT[N][K] (bf16), via LDS tile.
// block (32,8), grid (N/32, K/32)
// ---------------------------------------------------------------------------
__global__ void attn_wt_kernel(const float* __restrict__ W,
                               __bf16* __restrict__ WT, int K, int N) {
  __shared__ __bf16 tile[32][33];
  const int n0 = blockIdx.x * 32, k0 = blockIdx.y * 32;
  const int tx = threadIdx.x, ty = threadIdx.y;
#pragma unroll
  for (int i = 0; i < 32; i += 8)
    tile[ty + i][tx] = (__bf16)W[(size_t)(k0 + ty + i) * N + n0 + tx];
  __syncthreads();
#pragma unroll
  for (int i = 0; i < 32; i += 8)
    WT[(size_t)(n0 + ty + i) * K + k0 + tx] = tile[tx][ty + i];
}

// ---------------------------------------------------------------------------
// Fragment loaders (contiguous 16B chunks per the CDNA5 7.12.2 layouts)
// A (16x32): lane m = row, chunks at K = hi*8 and K = 16+hi*8
// B (32x16): lane m = col, chunks at K = hi*8 and K = 16+hi*8 (K-major source)
// ---------------------------------------------------------------------------
__device__ __forceinline__ v16bf frag_from2(const __bf16* p0, const __bf16* p1) {
  Frag16 f;
  f.h[0] = *(const bf16x8*)p0;
  f.h[1] = *(const bf16x8*)p1;
  return f.v;
}

// A fragment: row_base points at element (row m, K=0) of the 32-wide K slice
__device__ __forceinline__ v16bf afrag(const __bf16* row_base, int hi) {
  return frag_from2(row_base + hi * 8, row_base + 16 + hi * 8);
}
// B fragment: col_base points at element (col m, K=0) of a K-major row
__device__ __forceinline__ v16bf bfrag(const __bf16* col_base, int hi) {
  return frag_from2(col_base + hi * 8, col_base + 16 + hi * 8);
}

// ---------------------------------------------------------------------------
// QKV GEMM: X(4096x1024 bf16, row-major) @ WT(3072x1024 bf16, col-of-W-major)
// + bias -> scatter Q/K [B,H,N,64] (Q pre-scaled), V transposed [B,H,64,N].
// One wave computes a 32x32 output block (2x2 WMMA tiles).
// ---------------------------------------------------------------------------
__global__ void attn_qkv_gemm_kernel(const __bf16* __restrict__ X,
                                     const __bf16* __restrict__ WT,
                                     const float* __restrict__ bias,
                                     __bf16* __restrict__ Qb,
                                     __bf16* __restrict__ Kb,
                                     __bf16* __restrict__ Vt) {
  const int lane = threadIdx.x;
  const int m  = lane & 15;
  const int hi = lane >> 4;
  const int col0 = blockIdx.x * 32;
  const int row0 = blockIdx.y * 32;

  v8f acc[2][2] = {};
  for (int k0 = 0; k0 < DIM; k0 += 32) {
    const v16bf a0 = afrag(X + (size_t)(row0 + m) * DIM + k0, hi);
    const v16bf a1 = afrag(X + (size_t)(row0 + 16 + m) * DIM + k0, hi);
    const v16bf b0 = bfrag(WT + (size_t)(col0 + m) * DIM + k0, hi);
    const v16bf b1 = bfrag(WT + (size_t)(col0 + 16 + m) * DIM + k0, hi);
    acc[0][0] = WMMA_BF16(a0, b0, acc[0][0]);
    acc[0][1] = WMMA_BF16(a0, b1, acc[0][1]);
    acc[1][0] = WMMA_BF16(a1, b0, acc[1][0]);
    acc[1][1] = WMMA_BF16(a1, b1, acc[1][1]);
  }

#pragma unroll
  for (int j = 0; j < 2; ++j) {
    const int c = col0 + j * 16 + m;    // global column in [0,3072)
    const float bv = bias[c];
    const int sel = c >> 10;            // 0=Q 1=K 2=V
    const int hh  = (c >> 6) & 15;
    const int dd  = c & 63;
    const float sc = (sel == 0) ? QSCALE : 1.0f;
#pragma unroll
    for (int i = 0; i < 2; ++i) {
#pragma unroll
      for (int v = 0; v < 8; ++v) {
        const int g  = row0 + i * 16 + v + hi * 8;  // global row
        const int bb = g >> 11;
        const int n  = g & 2047;
        const int bh = bb * NH + hh;
        const float val = (acc[i][j][v] + bv) * sc;
        if (sel == 0)
          Qb[((size_t)bh * NSEQ + n) * HD + dd] = (__bf16)val;
        else if (sel == 1)
          Kb[((size_t)bh * NSEQ + n) * HD + dd] = (__bf16)val;
        else
          Vt[((size_t)bh * HD + dd) * NSEQ + n] = (__bf16)val;  // dim-major
      }
    }
  }
}

// ---------------------------------------------------------------------------
// Flash attention: one wave per (b, h, 16-row query tile).
// K stored [N,64] row-major (key-major => B frags contiguous for QK^T);
// V stored [64,N] dim-major (key-contiguous => B frags contiguous for P@V).
// ---------------------------------------------------------------------------
__global__ void attn_flash_kernel(const __bf16* __restrict__ Qb,
                                  const __bf16* __restrict__ Kb,
                                  const __bf16* __restrict__ Vt,
                                  __bf16* __restrict__ Ob) {
  const int lane = threadIdx.x;
  const int m  = lane & 15;
  const int hi = lane >> 4;
  const int q0 = blockIdx.x * 16;
  const int h  = blockIdx.y;
  const int b  = blockIdx.z;
  const int bh = b * NH + h;

  const __bf16* Qbase  = Qb + ((size_t)bh * NSEQ + q0) * HD;
  const __bf16* Kbase  = Kb + (size_t)bh * NSEQ * HD;
  const __bf16* Vtbase = Vt + (size_t)bh * HD * NSEQ;

  // Q A-fragments (dims 0..31 and 32..63), loaded once
  const v16bf qa0 = afrag(Qbase + (size_t)m * HD, hi);
  const v16bf qa1 = afrag(Qbase + (size_t)m * HD + 32, hi);

  v8f acc[4] = {};
  float mrow[8], lrow[8];
#pragma unroll
  for (int v = 0; v < 8; ++v) { mrow[v] = -INFINITY; lrow[v] = 0.0f; }

  __shared__ __bf16 sP[16][32];   // 16 rows x 32 keys, 64B row stride

  for (int kb0 = 0; kb0 < NSEQ; kb0 += 32) {
    // ---- S = Q @ K^T for 32 keys (two 16x16 tiles) ----
    const __bf16* kr0 = Kbase + (size_t)(kb0 + m) * HD;        // keys kb0..+15
    const __bf16* kr1 = Kbase + (size_t)(kb0 + 16 + m) * HD;   // keys +16..+31
    v8f s0 = {}, s1 = {};
    s0 = WMMA_BF16(qa0, bfrag(kr0, hi), s0);
    s0 = WMMA_BF16(qa1, bfrag(kr0 + 32, hi), s0);
    s1 = WMMA_BF16(qa0, bfrag(kr1, hi), s1);
    s1 = WMMA_BF16(qa1, bfrag(kr1 + 32, hi), s1);

    // ---- online softmax (rows live in disjoint lane halves) ----
#pragma unroll
    for (int v = 0; v < 8; ++v) {
      float smax = fmaxf(s0[v], s1[v]);
#pragma unroll
      for (int off = 8; off >= 1; off >>= 1)
        smax = fmaxf(smax, __shfl_xor(smax, off, 16));
      const float mnew  = fmaxf(mrow[v], smax);
      const float alpha = __expf(mrow[v] - mnew);
      const float p0 = __expf(s0[v] - mnew);
      const float p1 = __expf(s1[v] - mnew);
      float rs = p0 + p1;
#pragma unroll
      for (int off = 8; off >= 1; off >>= 1)
        rs += __shfl_xor(rs, off, 16);
      lrow[v] = lrow[v] * alpha + rs;
      mrow[v] = mnew;
#pragma unroll
      for (int t = 0; t < 4; ++t) acc[t][v] *= alpha;
      const int r = v + hi * 8;
      sP[r][m]      = (__bf16)p0;
      sP[r][16 + m] = (__bf16)p1;
    }
    __syncthreads();

    // ---- P as A fragment (two contiguous 16B LDS chunks per lane) ----
    const v16bf pa = frag_from2(&sP[m][hi * 8], &sP[m][16 + hi * 8]);

    // ---- O += P @ V (4 column tiles of 16 dims) ----
#pragma unroll
    for (int t = 0; t < 4; ++t) {
      const __bf16* vcol = Vtbase + (size_t)(t * 16 + m) * NSEQ + kb0;
      acc[t] = WMMA_BF16(pa, bfrag(vcol, hi), acc[t]);
    }
    __syncthreads();
  }

  // ---- normalize and store bf16 [B,N,DIM] with head interleave ----
#pragma unroll
  for (int v = 0; v < 8; ++v) {
    const float inv = 1.0f / lrow[v];
    const int n = q0 + v + hi * 8;
#pragma unroll
    for (int t = 0; t < 4; ++t) {
      const float val = acc[t][v] * inv;
      Ob[((size_t)(b * NSEQ + n)) * DIM + h * HD + t * 16 + m] = (__bf16)val;
    }
  }
}

// ---------------------------------------------------------------------------
// Proj GEMM: A(4096x1024 bf16) @ WT(1024x1024 bf16, col-of-W-major) + bias
// -> fp32 out. One wave computes a 32x32 block (2x2 WMMA tiles).
// ---------------------------------------------------------------------------
__global__ void attn_proj_gemm_kernel(const __bf16* __restrict__ A,
                                      const __bf16* __restrict__ WT,
                                      const float* __restrict__ bias,
                                      float* __restrict__ out) {
  const int lane = threadIdx.x;
  const int m  = lane & 15;
  const int hi = lane >> 4;
  const int col0 = blockIdx.x * 32;
  const int row0 = blockIdx.y * 32;

  v8f acc[2][2] = {};
  for (int k0 = 0; k0 < DIM; k0 += 32) {
    const v16bf a0 = afrag(A + (size_t)(row0 + m) * DIM + k0, hi);
    const v16bf a1 = afrag(A + (size_t)(row0 + 16 + m) * DIM + k0, hi);
    const v16bf b0 = bfrag(WT + (size_t)(col0 + m) * DIM + k0, hi);
    const v16bf b1 = bfrag(WT + (size_t)(col0 + 16 + m) * DIM + k0, hi);
    acc[0][0] = WMMA_BF16(a0, b0, acc[0][0]);
    acc[0][1] = WMMA_BF16(a0, b1, acc[0][1]);
    acc[1][0] = WMMA_BF16(a1, b0, acc[1][0]);
    acc[1][1] = WMMA_BF16(a1, b1, acc[1][1]);
  }

#pragma unroll
  for (int j = 0; j < 2; ++j) {
    const int c = col0 + j * 16 + m;
    const float bv = bias[c];
#pragma unroll
    for (int i = 0; i < 2; ++i) {
#pragma unroll
      for (int v = 0; v < 8; ++v) {
        const int g = row0 + i * 16 + v + hi * 8;
        out[(size_t)g * DIM + c] = acc[i][j][v] + bv;
      }
    }
  }
}

// ---------------------------------------------------------------------------
extern "C" void kernel_launch(void* const* d_in, const int* in_sizes, int n_in,
                              void* d_out, int out_size, void* d_ws, size_t ws_size,
                              hipStream_t stream) {
  const float* x      = (const float*)d_in[0];  // (B,N,DIM)
  const float* qkv_w  = (const float*)d_in[1];  // (DIM,3*DIM)
  const float* qkv_b  = (const float*)d_in[2];  // (3*DIM)
  const float* proj_w = (const float*)d_in[3];  // (DIM,DIM)
  const float* proj_b = (const float*)d_in[4];  // (DIM)
  float* out = (float*)d_out;

  // Workspace layout (bf16 buffers)
  char* ws = (char*)d_ws;
  const size_t SZ_X    = (size_t)ROWS * DIM * 2;              // 8 MB
  const size_t SZ_QKVW = (size_t)DIM * QKVC * 2;              // 6 MB
  const size_t SZ_PRJW = (size_t)DIM * DIM * 2;               // 2 MB
  const size_t SZ_HEAD = (size_t)BATCH * NH * NSEQ * HD * 2;  // 8 MB each
  __bf16* x_bf   = (__bf16*)(ws);
  __bf16* qwt_bf = (__bf16*)(ws + SZ_X);                      // [3072][1024]
  __bf16* pwt_bf = (__bf16*)(ws + SZ_X + SZ_QKVW);            // [1024][1024]
  __bf16* Qb     = (__bf16*)(ws + SZ_X + SZ_QKVW + SZ_PRJW);  // [B,H,N,64]
  __bf16* Kb     = (__bf16*)((char*)Qb + SZ_HEAD);            // [B,H,N,64]
  __bf16* Vt     = (__bf16*)((char*)Kb + SZ_HEAD);            // [B,H,64,N]
  __bf16* attn_o = (__bf16*)((char*)Vt + SZ_HEAD);            // [B,N,DIM]

  // 1) fp32 -> bf16 conversions (+ weight transposes to K-major)
  {
    const int nx = ROWS * DIM;
    attn_f2bf_kernel<<<(nx + 255) / 256, 256, 0, stream>>>(x, x_bf, nx);
    dim3 blk(32, 8);
    dim3 g1(QKVC / 32, DIM / 32);
    attn_wt_kernel<<<g1, blk, 0, stream>>>(qkv_w, qwt_bf, DIM, QKVC);
    dim3 g2(DIM / 32, DIM / 32);
    attn_wt_kernel<<<g2, blk, 0, stream>>>(proj_w, pwt_bf, DIM, DIM);
  }

  // 2) QKV GEMM + bias + scatter to per-head layouts (V dim-major)
  {
    dim3 grid(QKVC / 32, ROWS / 32);   // 96 x 128 waves
    attn_qkv_gemm_kernel<<<grid, 32, 0, stream>>>(x_bf, qwt_bf, qkv_b, Qb, Kb, Vt);
  }

  // 3) Flash attention per (b, h, 16-row q-tile)
  {
    dim3 grid(NSEQ / 16, NH, BATCH);   // 128 x 16 x 2 waves
    attn_flash_kernel<<<grid, 32, 0, stream>>>(Qb, Kb, Vt, attn_o);
  }

  // 4) Proj GEMM + bias -> fp32 output
  {
    dim3 grid(DIM / 32, ROWS / 32);    // 32 x 128 waves
    attn_proj_gemm_kernel<<<grid, 32, 0, stream>>>(attn_o, pwt_bf, proj_b, out);
  }
}
